// QKV_attention_65292092833994
// MI455X (gfx1250) — matrix operand order repaired and checked by reading, not compiled, not run on hardware
//
#include <hip/hip_runtime.h>

// ---------------- problem sizes ----------------
#define B_    16
#define TQ_   512
#define TK_   2048
#define D_    1024
#define ATTN_ 1024
#define F_    2048   // 2*D concat dim

// ---------------- tiling ----------------
#define QT        32     // query rows per workgroup
#define KT        64     // keys per flash iteration
#define NTHREADS  512    // 16 wave32 -> 4 waves per SIMD32 (one WG resident per WGP)

// ---------------- LDS strides (elements), multiples of 8 -> 16B-aligned rows ----
#define QST_ST  40      // QsT  [D_][QST_ST]  bf16, d-major Q
#define KS_ST   1032    // Ks   [KT][KS_ST]   bf16, row-major enc tile
#define CTXT_ST 40      // CtxT [D_][CTXT_ST] bf16, f-major context (overlays Ks)
#define SB_ST   36      // SbA/SbB [KT][SB_ST] f32, S^T partial scores (key rows, q cols)
#define PT_ST   72      // Pt   [QT][PT_ST]   bf16, probs row-major (q rows, key cols)

// ---------------- LDS layout (bytes) ----------------
#define OFF_QST 0
#define SZ_QST  (D_ * QST_ST * 2)            // 81920
#define OFF_RB  (OFF_QST + SZ_QST)
#define SZ_RB   (KT * KS_ST * 2)             // 132096 (Ks; reused by CtxT = 81920)
#define OFF_CTXT OFF_RB
#define OFF_SBA (OFF_RB + SZ_RB)             // 214016
#define SZ_SB   (KT * SB_ST * 4)             // 9216
#define OFF_SBB (OFF_SBA + SZ_SB)            // 223232
#define OFF_PT  (OFF_SBB + SZ_SB)            // 232448
#define SZ_PT   (QT * PT_ST * 2)             // 4608
#define OFF_RED (OFF_PT + SZ_PT)             // 237056
#define SZ_RED  (16 * QT * 4)                // 2048
#define OFF_ST  (OFF_RED + SZ_RED)           // 239104
#define SZ_ST   (3 * QT * 4)
#define SMEM_TOTAL (OFF_ST + SZ_ST)          // 239488 B ~ 234 KB < 320 KB

typedef __attribute__((ext_vector_type(16))) __bf16 v16bf;
typedef __attribute__((ext_vector_type(8)))  __bf16 v8bf;
typedef __attribute__((ext_vector_type(4)))  __bf16 v4bf;
typedef __attribute__((ext_vector_type(8)))  float  v8f;

__device__ __forceinline__ v8f zero8() {
  v8f z = {0.f, 0.f, 0.f, 0.f, 0.f, 0.f, 0.f, 0.f};
  return z;
}

// A operand, 16x32 bf16 from row-major LDS [row][k].
// lanes 0-15: M=lane, K lo-half (koff=0); lanes 16-31: M=lane-16, K hi-half (koff=8).
__device__ __forceinline__ v16bf ldsA(const __bf16* base, int stride, int row0, int k0, int lane) {
  const int m    = lane & 15;
  const int koff = (lane & 16) ? 8 : 0;
  const __bf16* p = base + (row0 + m) * stride + k0 + koff;
  v8bf lo = *(const v8bf*)p;
  v8bf hi = *(const v8bf*)(p + 16);
  v16bf r;
#pragma unroll
  for (int i = 0; i < 8; ++i) { r[i] = lo[i]; r[i + 8] = hi[i]; }
  return r;
}

// B operand, 32x16 bf16 from k-major LDS [k][n]: N = lane&15, K lo/hi-half (koff=0/8) like ldsA.
__device__ __forceinline__ v16bf ldsB(const __bf16* base, int stride, int k0, int n0, int lane) {
  const int nn   = lane & 15;
  const int koff = (lane & 16) ? 8 : 0;
  const __bf16* p = base + (k0 + koff) * stride + n0 + nn;
  v16bf r;
#pragma unroll
  for (int i = 0; i < 8; ++i) { r[i] = p[i * stride]; r[i + 8] = p[(16 + i) * stride]; }
  return r;
}

// A operand streamed straight from global f32 row-major [row][k], converted to bf16.
__device__ __forceinline__ v16bf ldgA(const float* base, int ld, int row0, int k0, int lane) {
  const int m    = lane & 15;
  const int koff = (lane & 16) ? 8 : 0;
  const float* p = base + (size_t)(row0 + m) * ld + k0 + koff;
  float4 x0 = *(const float4*)p;
  float4 x1 = *(const float4*)(p + 4);
  float4 x2 = *(const float4*)(p + 16);
  float4 x3 = *(const float4*)(p + 20);
  v16bf r;
  r[0]  = (__bf16)x0.x; r[1]  = (__bf16)x0.y; r[2]  = (__bf16)x0.z; r[3]  = (__bf16)x0.w;
  r[4]  = (__bf16)x1.x; r[5]  = (__bf16)x1.y; r[6]  = (__bf16)x1.z; r[7]  = (__bf16)x1.w;
  r[8]  = (__bf16)x2.x; r[9]  = (__bf16)x2.y; r[10] = (__bf16)x2.z; r[11] = (__bf16)x2.w;
  r[12] = (__bf16)x3.x; r[13] = (__bf16)x3.y; r[14] = (__bf16)x3.z; r[15] = (__bf16)x3.w;
  return r;
}

__device__ __forceinline__ v8f wmma_bf16(v16bf a, v16bf b, v8f c) {
  return __builtin_amdgcn_wmma_f32_16x16x32_bf16(false, a, false, b, (short)0, c, false, false);
}

__global__ __launch_bounds__(NTHREADS)
void fused_attn_kernel(const float* __restrict__ dec,
                       const float* __restrict__ enc,
                       const unsigned char* __restrict__ mask,
                       const float* __restrict__ W,
                       float* __restrict__ out) {
  __shared__ __align__(16) unsigned char smem[SMEM_TOTAL];
  __bf16* QsT  = (__bf16*)(smem + OFF_QST);  // [D_][QST_ST]   Q, d-major
  __bf16* Ks   = (__bf16*)(smem + OFF_RB);   // [KT][KS_ST]    enc tile, row-major
  __bf16* CtxT = (__bf16*)(smem + OFF_CTXT); // [D_][CTXT_ST]  context, f-major (phase 2)
  float*  SbA  = (float*)(smem + OFF_SBA);   // [KT][SB_ST]    S^T partial (D lower half)
  float*  SbB  = (float*)(smem + OFF_SBB);   // [KT][SB_ST]    S^T partial (D upper half)
  __bf16* Pt   = (__bf16*)(smem + OFF_PT);   // [QT][PT_ST]    probs, row-major
  float*  red  = (float*)(smem + OFF_RED);   // [16][QT]       partial max/sum
  float*  rowM = (float*)(smem + OFF_ST);
  float*  rowL = rowM + QT;
  float*  rowS = rowL + QT;

  const int tid  = threadIdx.x;
  const int lane = tid & 31;
  const int w    = tid >> 5;          // wave id 0..15
  const int bb   = blockIdx.y;
  const int q0   = blockIdx.x * QT;

  // ---- one-time: stage Q tile d-major: QsT[d][q] = bf16(dec[bb][q0+q][d]) ----
  for (int i = tid; i < QT * (D_ / 4); i += NTHREADS) {
    const int q  = i >> 8;
    const int c4 = i & 255;
    const float4 v = *(const float4*)(dec + ((size_t)bb * TQ_ + q0 + q) * D_ + c4 * 4);
    const int d = c4 * 4;
    QsT[(d + 0) * QST_ST + q] = (__bf16)v.x;
    QsT[(d + 1) * QST_ST + q] = (__bf16)v.y;
    QsT[(d + 2) * QST_ST + q] = (__bf16)v.z;
    QsT[(d + 3) * QST_ST + q] = (__bf16)v.w;
  }
  if (tid < QT) { rowM[tid] = -__builtin_inff(); rowL[tid] = 0.f; rowS[tid] = 1.f; }

  // context accumulators Ctx[q][d]: per wave 2 q-tiles x 4 d-tiles (d-tile = w + 16t)
  v8f ctx[2][4];
#pragma unroll
  for (int qi = 0; qi < 2; ++qi)
#pragma unroll
    for (int t = 0; t < 4; ++t) ctx[qi][t] = zero8();

  // S^T split: wave = (tile, D-half). 8 tiles of [64 keys x 32 q], 2 halves of D.
  const int half   = w & 1;
  const int stile  = w >> 1;          // 0..7
  const int kti    = stile >> 1;      // key sub-tile 0..3
  const int qti    = stile & 1;       // q   sub-tile 0..1
  const int dkbase = half * (D_ / 2);
  float* SbH = half ? SbB : SbA;

  const int n  = lane & 15;
  const int mb = (lane & 16) ? 8 : 0;

  // =================== flash loop over keys ===================
  for (int kt = 0; kt < TK_; kt += KT) {
    __syncthreads();  // (1) prev ctx WMMA done -> Ks/Sb/Pt reuse safe

    // stage enc tile row-major with packed stores: Ks[k][d]
    for (int i = tid; i < KT * (D_ / 4); i += NTHREADS) {
      const int kk = i >> 8;
      const int c4 = i & 255;
      const float4 v = *(const float4*)(enc + ((size_t)bb * TK_ + kt + kk) * D_ + c4 * 4);
      v4bf pk;
      pk[0] = (__bf16)v.x; pk[1] = (__bf16)v.y; pk[2] = (__bf16)v.z; pk[3] = (__bf16)v.w;
      *(v4bf*)(Ks + kk * KS_ST + c4 * 4) = pk;
    }
    __syncthreads();  // (2)

    // prefetch next enc tile into cache while S^T WMMA runs (global_prefetch_b8)
    if (kt + KT < TK_) {
      const float* nx = enc + ((size_t)bb * TK_ + kt + KT) * D_;
      for (int i = tid; i < KT * 32; i += NTHREADS)        // 128B lines
        __builtin_prefetch(nx + (size_t)(i >> 5) * D_ + (i & 31) * 32, 0, 3);
    }

    // ---- S^T = K · Q^T : wave owns one tile x one D-half ----
    v8f s = zero8();
#pragma unroll 4
    for (int dk = 0; dk < D_ / 2; dk += 32) {
      v16bf a = ldsA(Ks, KS_ST, kti * 16, dkbase + dk, lane);     // A: enc rows over d
      v16bf b = ldsB(QsT, QST_ST, dkbase + dk, qti * 16, lane);   // B: Q^T (lane=d)
      s = wmma_bf16(a, b, s);
    }
#pragma unroll
    for (int v = 0; v < 8; ++v)
      SbH[(kti * 16 + mb + v) * SB_ST + qti * 16 + n] = s[v];
    __syncthreads();  // (3)

    // ---- parallel online softmax: 512 threads -> (q = tid&31, 4-key segment) ----
    const int sq  = tid & 31;
    const int seg = tid >> 5;   // 0..15
    const unsigned int m4 =
        *(const unsigned int*)(mask + ((size_t)bb * TQ_ + q0 + sq) * TK_ + kt + seg * 4);
    float sv[4];
    float pmax = -__builtin_inff();
#pragma unroll
    for (int j = 0; j < 4; ++j) {
      const int key = seg * 4 + j;
      sv[j] = SbA[key * SB_ST + sq] + SbB[key * SB_ST + sq];   // combine D halves
      if (!((m4 >> (8 * j)) & 0xff)) pmax = fmaxf(pmax, sv[j]);
    }
    red[seg * QT + sq] = pmax;
    __syncthreads();  // (4)

    if (tid < QT) {
      const int q = tid;
      const float mold = rowM[q];
      float mx = mold;
#pragma unroll
      for (int s2 = 0; s2 < 16; ++s2) mx = fmaxf(mx, red[s2 * QT + q]);
      rowS[q] = (mx == mold) ? 1.f : __expf(mold - mx);
      rowM[q] = mx;
    }
    __syncthreads();  // (5)

    {
      const float mx = rowM[sq];
      float psum = 0.f;
      v4bf pk;
#pragma unroll
      for (int j = 0; j < 4; ++j) {
        const float e = ((m4 >> (8 * j)) & 0xff) ? 0.f : __expf(sv[j] - mx);
        psum += e;
        pk[j] = (__bf16)e;
      }
      *(v4bf*)(Pt + sq * PT_ST + seg * 4) = pk;
      red[seg * QT + sq] = psum;
    }
    __syncthreads();  // (6)

    // ---- rowL update folded in (wave 0 side-work), rescale, context WMMA ----
    if (tid < QT) {
      float acc = rowL[tid] * rowS[tid];
#pragma unroll
      for (int s2 = 0; s2 < 16; ++s2) acc += red[s2 * QT + tid];
      rowL[tid] = acc;
    }
    {
      float sc[2][8];
#pragma unroll
      for (int qi = 0; qi < 2; ++qi)
#pragma unroll
        for (int v = 0; v < 8; ++v) sc[qi][v] = rowS[qi * 16 + mb + v];
#pragma unroll
      for (int qi = 0; qi < 2; ++qi)
#pragma unroll
        for (int t = 0; t < 4; ++t)
#pragma unroll
          for (int e = 0; e < 8; ++e) ctx[qi][t][e] *= sc[qi][e];
    }
    // Ctx[q][d] += P · V : A = Pt rows (q over key), B = Ks (lane=key, cols=d)
#pragma unroll
    for (int kc = 0; kc < 2; ++kc) {
      const int kk0 = kc * 32;
      v16bf a0 = ldsA(Pt, PT_ST, 0,  kk0, lane);
      v16bf a1 = ldsA(Pt, PT_ST, 16, kk0, lane);
#pragma unroll
      for (int t = 0; t < 4; ++t) {
        v16bf b = ldsB(Ks, KS_ST, kk0, (w + 16 * t) * 16, lane);
        ctx[0][t] = wmma_bf16(a0, b, ctx[0][t]);
        ctx[1][t] = wmma_bf16(a1, b, ctx[1][t]);
      }
    }
  }
  __syncthreads();  // all context WMMA done; Ks region free

  // ---- normalize and store context f-major: CtxT[d][q] (packed v8bf stores) ----
  {
    float iv[2][8];
#pragma unroll
    for (int qi = 0; qi < 2; ++qi)
#pragma unroll
      for (int v = 0; v < 8; ++v) iv[qi][v] = 1.f / rowL[qi * 16 + mb + v];
#pragma unroll
    for (int qi = 0; qi < 2; ++qi)
#pragma unroll
      for (int t = 0; t < 4; ++t) {
        v8bf pk;
#pragma unroll
        for (int e = 0; e < 8; ++e) pk[e] = (__bf16)(ctx[qi][t][e] * iv[qi][e]);
        *(v8bf*)(CtxT + ((w + 16 * t) * 16 + n) * CTXT_ST + qi * 16 + mb) = pk;
      }
  }
  __syncthreads();

  // =================== phase 2: out^T = W · cat^T, W streamed from global ===================
  v8f oacc[2][4];
#pragma unroll
  for (int qi = 0; qi < 2; ++qi)
#pragma unroll
    for (int t = 0; t < 4; ++t) oacc[qi][t] = zero8();

  for (int fc = 0; fc < F_; fc += 32) {
    const __bf16* bt = (fc < D_) ? CtxT : QsT;      // cat = [context ; dec], f-major
    const int fk = (fc < D_) ? fc : fc - D_;
    v16bf b0 = ldsB(bt, 40, fk, 0, lane);           // CTXT_ST == QST_ST == 40
    v16bf b1 = ldsB(bt, 40, fk, 16, lane);
#pragma unroll
    for (int t = 0; t < 4; ++t) {
      v16bf a = ldgA(W, F_, (w + 16 * t) * 16, fc, lane);  // W rows over f, f32->bf16 in regs
      oacc[0][t] = wmma_bf16(a, b0, oacc[0][t]);
      oacc[1][t] = wmma_bf16(a, b1, oacc[1][t]);
    }
  }

  // ---- store out^T tiles as packed float4 pairs: out[b][q][a] ----
#pragma unroll
  for (int qi = 0; qi < 2; ++qi)
#pragma unroll
    for (int t = 0; t < 4; ++t) {
      const int q = q0 + qi * 16 + n;
      const int abase = (w + 16 * t) * 16 + mb;
      float* op = out + ((size_t)bb * TQ_ + q) * ATTN_ + abase;
      float4 lo = make_float4(oacc[qi][t][0], oacc[qi][t][1], oacc[qi][t][2], oacc[qi][t][3]);
      float4 hi = make_float4(oacc[qi][t][4], oacc[qi][t][5], oacc[qi][t][6], oacc[qi][t][7]);
      *(float4*)op = lo;
      *(float4*)(op + 4) = hi;
    }
}

extern "C" void kernel_launch(void* const* d_in, const int* in_sizes, int n_in,
                              void* d_out, int out_size, void* d_ws, size_t ws_size,
                              hipStream_t stream) {
  (void)in_sizes; (void)n_in; (void)out_size; (void)d_ws; (void)ws_size;
  const float*         dec  = (const float*)d_in[0];          // [16,512,1024] f32
  const float*         enc  = (const float*)d_in[1];          // [16,2048,1024] f32
  const unsigned char* mask = (const unsigned char*)d_in[2];  // [16,512,2048] bool (1B)
  const float*         W    = (const float*)d_in[3];          // [1024,2048] f32
  float*               out  = (float*)d_out;                  // [16,512,1024] f32

  dim3 grid(TQ_ / QT, B_);   // 16 x 16 = 256 workgroups, 512 threads each
  fused_attn_kernel<<<grid, NTHREADS, 0, stream>>>(dec, enc, mask, W, out);
}
